// Mamba2_9363028706195
// MI455X (gfx1250) — matrix operand order repaired and checked by reading, not compiled, not run on hardware
//
#include <hip/hip_runtime.h>
#include <hip/hip_bf16.h>
#include <cstdint>

#ifndef __has_builtin
#define __has_builtin(x) 0
#endif

#define D_MODEL   1024
#define D_STATE   128
#define D_CONV    4
#define HEADDIM   64
#define CHUNK     64
#define D_INNER   2048
#define NHEADS    32
#define D_IN_PROJ 4384   // 2*D_INNER + 2*D_STATE + NHEADS
#define CONV_DIM  2304   // D_INNER + 2*D_STATE
#define B_SZ      2
#define L_SEQ     4096
#define NROWS     (B_SZ * L_SEQ)   // 8192
#define NCHUNK    (L_SEQ / CHUNK)  // 64
#define N1PAD     4480             // ceil(4384/128)*128

typedef __attribute__((ext_vector_type(16))) _Float16 v16h;
typedef __attribute__((ext_vector_type(8)))  float    v8f;
typedef __attribute__((ext_vector_type(4)))  int      v4i_t;

// ---------------------------------------------------------------------------
// Async global->LDS staging (gfx1250 GLOBAL_LOAD_ASYNC_TO_LDS_B128, ASYNCcnt),
// guarded so the file also compiles on toolchains without the builtins.
// Builtin signature (from compiler diagnostic): (v4i*, v4i*, imm, imm).
// ---------------------------------------------------------------------------
#if __has_builtin(__builtin_amdgcn_global_load_async_to_lds_b128) && \
    __has_builtin(__builtin_amdgcn_s_wait_asynccnt)
#define HAVE_ASYNC_LDS 1
__device__ __forceinline__ void async_copy16(const void* g, void* l) {
    __builtin_amdgcn_global_load_async_to_lds_b128((v4i_t*)g, (v4i_t*)l, 0, 0);
}
__device__ __forceinline__ void async_wait0() {
    __builtin_amdgcn_s_wait_asynccnt(0);
}
#else
__device__ __forceinline__ void async_copy16(const void* g, void* l) {
    *(uint4*)l = *(const uint4*)g;
}
__device__ __forceinline__ void async_wait0() {}
#endif

// ---------------------------------------------------------------------------
// WMMA fragment loaders for v_wmma_f32_16x16x32_f16 (wave32), contiguous K order (same in A and B).
// A (16x32 f16): lane -> M=lane&15, K halves kb..kb+15,
//                kb = 0 (lanes 0-15) / 16 (lanes 16-31). Source: [M][ldk].
// B (32x16 f16): lane -> N=lane&15, K halves kb..kb+15, kb = 0 / 16.
//                Source stored transposed: [N][ldk].
// C/D (16x16 f32, 8 VGPRs): row = ((lane>>4)<<3)+r, col = lane&15.
// ---------------------------------------------------------------------------
__device__ __forceinline__ v16h load_frag_a(const _Float16* base, int ldk, int lane) {
    v16h a;
    unsigned int* au = (unsigned int*)&a;
    const int r  = lane & 15;
    const int kb = (lane >> 4) << 4;
    const _Float16* rowp = base + r * ldk;
#pragma unroll
    for (int p = 0; p < 8; ++p)
        au[p] = *(const unsigned int*)(rowp + kb + 2 * p);
    return a;
}

__device__ __forceinline__ v16h load_frag_b(const _Float16* base, int ldk, int lane) {
    v16h b;
    unsigned int* bu = (unsigned int*)&b;
    const int c  = lane & 15;
    const int kb = (lane >> 4) << 4;
    const _Float16* rowp = base + c * ldk + kb;
#pragma unroll
    for (int p = 0; p < 8; ++p)
        bu[p] = *(const unsigned int*)(rowp + 2 * p);
    return b;
}

__device__ __forceinline__ v8f wmma_f16(v16h a, v16h b, v8f c) {
    return __builtin_amdgcn_wmma_f32_16x16x32_f16(false, a, false, b, (short)0, c, false, false);
}

__device__ __forceinline__ float silu_f(float v) { return v / (1.f + __expf(-v)); }

// ---------------------------------------------------------------------------
// Pre-passes: f32 -> f16 convert, and f32 -> f16 transpose (W[K][N] -> WT[N][K],
// zero-padded to NPAD rows so GEMM staging needs no N guard).
// ---------------------------------------------------------------------------
__global__ void f32_to_f16_kernel(const float* __restrict__ src,
                                  _Float16* __restrict__ dst, int n) {
    const int i = blockIdx.x * 256 + threadIdx.x;
    if (i < n) dst[i] = (_Float16)src[i];
}

__global__ void transpose_f16_kernel(const float* __restrict__ W,
                                     _Float16* __restrict__ WT,
                                     int K, int N, int NPAD) {
    const int i = blockIdx.x * 256 + threadIdx.x;
    if (i >= NPAD * K) return;
    const int n = i / K, k = i % K;
    WT[i] = (n < N) ? (_Float16)W[(size_t)k * N + n] : (_Float16)0.f;
}

// ---------------------------------------------------------------------------
// f16 GEMM with f32 accumulate: C[M][N] = A16[M][K] * WT16[N][K]^T.
// Block = 256 threads (8 waves); tile 128x128, BK=32.  Tiles staged to LDS
// via async b128 copies; each wave owns a 16-row M strip and 8 WMMA frags.
// M, K multiples of 128/32; WT padded so only the C store is N-guarded.
// ---------------------------------------------------------------------------
__global__ void __launch_bounds__(256)
wmma_gemm16_kernel(const _Float16* __restrict__ A, const _Float16* __restrict__ WT,
                   float* __restrict__ C, int M, int N, int K) {
    const int tid  = threadIdx.x;
    const int lane = tid & 31;
    const int wave = tid >> 5;
    const int m0   = blockIdx.y * 128;
    const int n0   = blockIdx.x * 128;

    __shared__ _Float16 sA[128][40];   // [M][K] rows padded to 80B (16B aligned)
    __shared__ _Float16 sW[128][40];   // [N][K]

    v8f acc[8] = {};

    for (int k0 = 0; k0 < K; k0 += 32) {
        // stage A tile: 128 rows x 64B = 512 b128 transfers
        for (int i = tid; i < 128 * 4; i += 256) {
            const int m = i >> 2, q = (i & 3) * 8;
            async_copy16(A + (size_t)(m0 + m) * K + k0 + q, &sA[m][q]);
        }
        // stage W tile (already transposed/padded): 512 b128 transfers
        for (int i = tid; i < 128 * 4; i += 256) {
            const int n = i >> 2, q = (i & 3) * 8;
            async_copy16(WT + (size_t)(n0 + n) * K + k0 + q, &sW[n][q]);
        }
        async_wait0();
        __syncthreads();

        const v16h af = load_frag_a(&sA[wave * 16][0], 40, lane);
#pragma unroll
        for (int t = 0; t < 8; ++t) {
            const v16h bf = load_frag_b(&sW[t * 16][0], 40, lane);
            acc[t] = wmma_f16(af, bf, acc[t]);
        }
        __syncthreads();
    }

    const int mo = m0 + wave * 16 + ((lane >> 4) << 3);
#pragma unroll
    for (int t = 0; t < 8; ++t) {
        const int n = n0 + t * 16 + (lane & 15);
        if (n < N) {
#pragma unroll
            for (int r = 0; r < 8; ++r)
                C[(size_t)(mo + r) * N + n] = acc[t][r];
        }
    }
}

// ---------------------------------------------------------------------------
// dt = softplus(zxbcdt[..., -NHEADS:] + dt_bias)
// ---------------------------------------------------------------------------
__global__ void dt_kernel(const float* __restrict__ zxbcdt,
                          const float* __restrict__ dt_bias,
                          float* __restrict__ dtbuf) {
    const int idx = blockIdx.x * 256 + threadIdx.x;
    if (idx >= NROWS * NHEADS) return;
    const int h   = idx % NHEADS;
    const int row = idx / NHEADS;
    const float v = zxbcdt[(size_t)row * D_IN_PROJ + (D_IN_PROJ - NHEADS) + h] + dt_bias[h];
    dtbuf[idx] = (v > 20.f) ? v : log1pf(__expf(v));
}

// ---------------------------------------------------------------------------
// Depthwise causal conv (width 4) over L, + bias, + SiLU.
// ---------------------------------------------------------------------------
__global__ void conv_silu_kernel(const float* __restrict__ zxbcdt,
                                 const float* __restrict__ conv_w,
                                 const float* __restrict__ conv_b,
                                 float* __restrict__ xBCc) {
    const size_t idx = (size_t)blockIdx.x * 256 + threadIdx.x;
    if (idx >= (size_t)NROWS * CONV_DIM) return;
    const int    c   = (int)(idx % CONV_DIM);
    const size_t row = idx / CONV_DIM;
    const int    l   = (int)(row % L_SEQ);
    float acc = conv_b[c];
#pragma unroll
    for (int j = 0; j < D_CONV; ++j) {
        const int ll = l - (D_CONV - 1) + j;
        if (ll >= 0)
            acc += conv_w[c * D_CONV + j] *
                   zxbcdt[(row - (size_t)(D_CONV - 1 - j)) * D_IN_PROJ + D_INNER + c];
    }
    xBCc[idx] = silu_f(acc);
}

// ---------------------------------------------------------------------------
// SSD chunked scan.  One block (128 thr / 4 waves) per (head, batch).
// Sequential over 64 chunks; 64x128 f32 state carried in LDS.
// All 4 einsums done with v_wmma_f32_16x16x32_f16.
// ---------------------------------------------------------------------------
__global__ void __launch_bounds__(128)
ssd_kernel(const float* __restrict__ xBCc, const float* __restrict__ dtbuf,
           const float* __restrict__ A_log, const float* __restrict__ Dv,
           float* __restrict__ yh) {
    const int h    = blockIdx.x;
    const int b    = blockIdx.y;
    const int tid  = threadIdx.x;
    const int lane = tid & 31;
    const int wave = tid >> 5;

    __shared__ float    S[HEADDIM][D_STATE];      // running state, f32 master
    __shared__ _Float16 sS16[HEADDIM][D_STATE];   // f16 copy (B-operand [p][n])
    __shared__ _Float16 sC [CHUNK][D_STATE];      // C          (A-op [l][n])
    __shared__ _Float16 sCd[CHUNK][D_STATE];      // C*exp(cum) (A-op [l][n])
    __shared__ _Float16 sB [CHUNK][D_STATE];      // B          (B-op [s][n])
    __shared__ _Float16 sBd[D_STATE][CHUNK];      // B*decay    (B-op [n][l])
    __shared__ _Float16 sXdT[HEADDIM][CHUNK];     // x*dt, [p][l] (A-op & B-op)
    __shared__ _Float16 sG [CHUNK][CHUNK];        // masked C·Bᵀ (A-op [l][s])
    __shared__ float    cum[CHUNK];
    __shared__ float    dtv[CHUNK];

    const float Ah = -__expf(A_log[h]);
    const float Dh = Dv[h];

    for (int i = tid; i < HEADDIM * D_STATE; i += 128) {
        ((float*)S)[i]       = 0.f;
        ((_Float16*)sS16)[i] = (_Float16)0.f;
    }
    __syncthreads();

    for (int c = 0; c < NCHUNK; ++c) {
        const int l0 = b * L_SEQ + c * CHUNK;

        if (tid < CHUNK)
            dtv[tid] = dtbuf[(size_t)(l0 + tid) * NHEADS + h];
        __syncthreads();
        if (tid == 0) {
            float s = 0.f;
            for (int l = 0; l < CHUNK; ++l) { s += Ah * dtv[l]; cum[l] = s; }
        }
        __syncthreads();
        const float cumlast = cum[CHUNK - 1];

        for (int i = tid; i < CHUNK * D_STATE; i += 128) {
            const int l = i >> 7, n = i & 127;
            const size_t row = (size_t)(l0 + l) * CONV_DIM;
            const float Bf = xBCc[row + D_INNER + n];
            const float Cf = xBCc[row + D_INNER + D_STATE + n];
            sB [l][n] = (_Float16)Bf;
            sBd[n][l] = (_Float16)(Bf * __expf(cumlast - cum[l]));
            sC [l][n] = (_Float16)Cf;
            sCd[l][n] = (_Float16)(Cf * __expf(cum[l]));
        }
        for (int i = tid; i < CHUNK * HEADDIM; i += 128) {
            const int l = i >> 6, p = i & 63;
            const float xv = xBCc[(size_t)(l0 + l) * CONV_DIM + h * HEADDIM + p];
            sXdT[p][l] = (_Float16)(xv * dtv[l]);
        }
        __syncthreads();

        // ---- G = C · Bᵀ (64x64, K=128), apply causal decay mask -> sG ----
        {
            const int m0 = wave * 16;
#pragma unroll
            for (int t = 0; t < 4; ++t) {
                v8f acc = {};
#pragma unroll
                for (int ks = 0; ks < 4; ++ks) {
                    const v16h af = load_frag_a(&sC[m0][ks * 32], D_STATE, lane);
                    const v16h bf = load_frag_b(&sB[t * 16][ks * 32], D_STATE, lane);
                    acc = wmma_f16(af, bf, acc);
                }
                const int mo = m0 + ((lane >> 4) << 3);
                const int n  = t * 16 + (lane & 15);
#pragma unroll
                for (int r = 0; r < 8; ++r) {
                    const int m = mo + r;
                    const float v = (n <= m) ? acc[r] * __expf(cum[m] - cum[n]) : 0.f;
                    sG[m][n] = (_Float16)v;
                }
            }
        }
        __syncthreads();

        // ---- Y = Gm · xd  +  Cdecay · S_oldᵀ ----
        v8f Y[4];
        {
            const int m0 = wave * 16;
#pragma unroll
            for (int t = 0; t < 4; ++t) {
                v8f acc = {};
#pragma unroll
                for (int ks = 0; ks < 2; ++ks) {
                    const v16h af = load_frag_a(&sG[m0][ks * 32], CHUNK, lane);
                    const v16h bf = load_frag_b(&sXdT[t * 16][ks * 32], CHUNK, lane);
                    acc = wmma_f16(af, bf, acc);
                }
#pragma unroll
                for (int ks = 0; ks < 4; ++ks) {
                    const v16h af = load_frag_a(&sCd[m0][ks * 32], D_STATE, lane);
                    const v16h bf = load_frag_b(&sS16[t * 16][ks * 32], D_STATE, lane);
                    acc = wmma_f16(af, bf, acc);
                }
                Y[t] = acc;
            }
        }
        __syncthreads();   // all reads of old sS16 complete before update

        // ---- states_c = xdᵀ · Bdecay; S = exp(cumlast)*S + states_c ----
        {
            const int m0   = wave * 16;
            const float df = __expf(cumlast);
#pragma unroll
            for (int nt = 0; nt < 8; ++nt) {
                v8f acc = {};
#pragma unroll
                for (int ks = 0; ks < 2; ++ks) {
                    const v16h af = load_frag_a(&sXdT[m0][ks * 32], CHUNK, lane);
                    const v16h bf = load_frag_b(&sBd[nt * 16][ks * 32], CHUNK, lane);
                    acc = wmma_f16(af, bf, acc);
                }
                const int po = m0 + ((lane >> 4) << 3);
                const int n  = nt * 16 + (lane & 15);
#pragma unroll
                for (int r = 0; r < 8; ++r) {
                    const float s = S[po + r][n] * df + acc[r];
                    S[po + r][n]    = s;
                    sS16[po + r][n] = (_Float16)s;
                }
            }
        }

        // ---- epilogue: y = Y + x_raw * D[h] ----
        {
            const int m0 = wave * 16;
#pragma unroll
            for (int t = 0; t < 4; ++t) {
                const int mo = m0 + ((lane >> 4) << 3);
                const int p  = t * 16 + (lane & 15);
#pragma unroll
                for (int r = 0; r < 8; ++r) {
                    const int l = mo + r;
                    const float xv = xBCc[(size_t)(l0 + l) * CONV_DIM + h * HEADDIM + p];
                    yh[((size_t)(l0 + l) * NHEADS + h) * HEADDIM + p] = Y[t][r] + xv * Dh;
                }
            }
        }
        __syncthreads();
    }
}

// ---------------------------------------------------------------------------
// y = yh * silu(z); RMSNorm over D_INNER; * norm_w.  Writes f16 for GEMM2.
// ---------------------------------------------------------------------------
__global__ void __launch_bounds__(256)
gate_norm_kernel(const float* __restrict__ zxbcdt, const float* __restrict__ yh,
                 const float* __restrict__ norm_w, _Float16* __restrict__ yn16) {
    const int row = blockIdx.x;
    __shared__ float red[256];
    const float* z = zxbcdt + (size_t)row * D_IN_PROJ;
    const float* y = yh + (size_t)row * D_INNER;

    float vals[8];
    float ss = 0.f;
#pragma unroll
    for (int i = 0; i < 8; ++i) {
        const int d = threadIdx.x + i * 256;
        const float v = y[d] * silu_f(z[d]);
        vals[i] = v;
        ss += v * v;
    }
    red[threadIdx.x] = ss;
    __syncthreads();
    for (int s = 128; s > 0; s >>= 1) {
        if (threadIdx.x < s) red[threadIdx.x] += red[threadIdx.x + s];
        __syncthreads();
    }
    const float r = rsqrtf(red[0] / (float)D_INNER + 1e-5f);
#pragma unroll
    for (int i = 0; i < 8; ++i) {
        const int d = threadIdx.x + i * 256;
        yn16[(size_t)row * D_INNER + d] = (_Float16)(vals[i] * r * norm_w[d]);
    }
}

// ---------------------------------------------------------------------------
extern "C" void kernel_launch(void* const* d_in, const int* in_sizes, int n_in,
                              void* d_out, int out_size, void* d_ws, size_t ws_size,
                              hipStream_t stream) {
    (void)in_sizes; (void)n_in; (void)out_size; (void)ws_size;

    const float* u          = (const float*)d_in[0];
    const float* in_proj_w  = (const float*)d_in[1];
    const float* conv_w     = (const float*)d_in[2];
    const float* conv_b     = (const float*)d_in[3];
    const float* dt_bias    = (const float*)d_in[4];
    const float* A_log      = (const float*)d_in[5];
    const float* Dv         = (const float*)d_in[6];
    const float* norm_w     = (const float*)d_in[7];
    const float* out_proj_w = (const float*)d_in[8];
    float*       out        = (float*)d_out;

    char* ws = (char*)d_ws;
    size_t off = 0;
    float* zxbcdt = (float*)(ws + off);  off += (size_t)NROWS * D_IN_PROJ * 4;   // 143.7 MB
    float* xBCc   = (float*)(ws + off);  off += (size_t)NROWS * CONV_DIM * 4;    // 75.5 MB
    float* dtbuf  = (float*)(ws + off);  off += (size_t)NROWS * NHEADS * 4;      // 1 MB
    float* yh     = (float*)(ws + off);  off += (size_t)NROWS * D_INNER * 4;     // 67.1 MB
    _Float16* u16 = (_Float16*)(ws + off); off += (size_t)NROWS * D_MODEL * 2;   // 16.8 MB
    _Float16* w1T = (_Float16*)(ws + off); off += (size_t)N1PAD * D_MODEL * 2;   //  9.2 MB
    _Float16* w2T = (_Float16*)(ws + off); off += (size_t)D_MODEL * D_INNER * 2; //  4.2 MB
    _Float16* yn16 = (_Float16*)xBCc;    // overlay: conv buffer dead after SSD

    // 0) f16 pre-passes (one-time converts / transposes)
    f32_to_f16_kernel<<<(NROWS * D_MODEL + 255) / 256, 256, 0, stream>>>(
        u, u16, NROWS * D_MODEL);
    transpose_f16_kernel<<<(N1PAD * D_MODEL + 255) / 256, 256, 0, stream>>>(
        in_proj_w, w1T, D_MODEL, D_IN_PROJ, N1PAD);
    transpose_f16_kernel<<<(D_MODEL * D_INNER + 255) / 256, 256, 0, stream>>>(
        out_proj_w, w2T, D_INNER, D_MODEL, D_MODEL);

    // 1) zxbcdt = u @ in_proj_w    (M=8192, N=4384, K=1024)
    wmma_gemm16_kernel<<<dim3(N1PAD / 128, NROWS / 128), 256, 0, stream>>>(
        u16, w1T, zxbcdt, NROWS, D_IN_PROJ, D_MODEL);

    // 2) dt = softplus(... + dt_bias)
    dt_kernel<<<(NROWS * NHEADS + 255) / 256, 256, 0, stream>>>(zxbcdt, dt_bias, dtbuf);

    // 3) depthwise causal conv + SiLU
    conv_silu_kernel<<<(int)(((size_t)NROWS * CONV_DIM + 255) / 256), 256, 0, stream>>>(
        zxbcdt, conv_w, conv_b, xBCc);

    // 4) SSD chunked scan (one block per (head, batch))
    ssd_kernel<<<dim3(NHEADS, B_SZ), 128, 0, stream>>>(xBCc, dtbuf, A_log, Dv, yh);

    // 5) gate with silu(z), RMSNorm -> f16
    gate_norm_kernel<<<NROWS, 256, 0, stream>>>(zxbcdt, yh, norm_w, yn16);

    // 6) out = ynorm @ out_proj_w   (M=8192, N=1024, K=2048)
    wmma_gemm16_kernel<<<dim3(D_MODEL / 128, NROWS / 128), 256, 0, stream>>>(
        yn16, w2T, out, NROWS, D_MODEL, D_INNER);
}
